// Contextual_Loss_4818953306342
// MI455X (gfx1250) — matrix operand order repaired and checked by reading, not compiled, not run on hardware
//
#include <hip/hip_runtime.h>
#include <hip/hip_bf16.h>
#include <math.h>

// ---------------------------------------------------------------------------
// Contextual loss for MI455X (gfx1250, wave32, WMMA).
// cos = I^T T is a 9216x9216x256 GEMM done with v_wmma_f32_16x16x32_f16.
// Features (f16, 9.4MB) stay L2-resident.  Each GEMM pass: 8 waves/block hold
// 8 distinct 16-row register fragments; the streamed operand is staged
// tile-by-tile into double-buffered LDS (async global->LDS when available)
// and consumed via bank-conflict-free ds_load_b128 fragment reads.
// ---------------------------------------------------------------------------

#define C_DIM 256                 // channels (K of the GEMM)
#define S_DIM 9216                // spatial positions (96*96)
#define NT    576                 // S_DIM / 16
#define KT    8                   // C_DIM / 32 (WMMA k-steps per tile)
#define NB    72                  // S_DIM / 128 (macro-tiles of 8 waves)
#define JCH   8                   // grid.y streaming chunks
#define LDS_STRIDE 264            // halves per staged row (256 + 8 pad = 528B)
#define TILE_H (16 * 256)         // halves per 16-row feature tile in global
#define LDS_TILE_H (16 * LDS_STRIDE)

#if defined(__has_builtin)
#if __has_builtin(__builtin_amdgcn_global_load_async_to_lds_b128)
#define HAVE_ASYNC_LDS 1
#endif
#endif

typedef _Float16 half8  __attribute__((ext_vector_type(8)));
typedef _Float16 half16 __attribute__((ext_vector_type(16)));
typedef float    float8 __attribute__((ext_vector_type(8)));

union FragU { half16 v; half8 h[2]; };

// ---- fragment loads (ISA 7.12.2 lane layouts) ------------------------------
// A (16x32 f16): lane<16 row M=lane, halves0-7=K k0..k0+7, halves8-15=K+16..+23;
// lane>=16 same rows, K ranges shifted by +8.
__device__ __forceinline__ half16 load_a(const _Float16* __restrict__ F,
                                         int row0, int k0, int lane) {
  int m = lane & 15, kh = lane >> 4;
  const _Float16* p = F + (size_t)(row0 + m) * C_DIM + k0 + (kh << 3);
  FragU f;
  f.h[0] = *(const half8*)(p);
  f.h[1] = *(const half8*)(p + 16);
  return f.v;
}
// B (32x16 f16): column N=lane&15, same K layout as A (halves0-7=K k0..k0+7, 8-15=K+16..+23; lane>=16 shifted +8).
__device__ __forceinline__ half16 load_b(const _Float16* __restrict__ F,
                                         int col0, int k0, int lane) {
  int n = lane & 15, kh = lane >> 4;
  const _Float16* p = F + (size_t)(col0 + n) * C_DIM + k0 + (kh << 3);
  FragU f;
  f.h[0] = *(const half8*)(p);
  f.h[1] = *(const half8*)(p + 16);
  return f.v;
}
// Same fragments out of a padded LDS tile (row stride 264 halves: lane n's 16B
// read starts at dword 4n mod 64 -> all 64 banks touched exactly once).
__device__ __forceinline__ half16 lds_a(const _Float16* __restrict__ buf,
                                        int k0, int lane) {
  int m = lane & 15, kh = lane >> 4;
  const _Float16* p = buf + m * LDS_STRIDE + k0 + (kh << 3);
  FragU f;
  f.h[0] = *(const half8*)(p);
  f.h[1] = *(const half8*)(p + 16);
  return f.v;
}
__device__ __forceinline__ half16 lds_b(const _Float16* __restrict__ buf,
                                        int k0, int lane) {
  int n = lane & 15, kh = lane >> 4;
  const _Float16* p = buf + n * LDS_STRIDE + k0 + (kh << 3);
  FragU f;
  f.h[0] = *(const half8*)(p);
  f.h[1] = *(const half8*)(p + 16);
  return f.v;
}

__device__ __forceinline__ float8 zero8() {
  float8 z;
#pragma unroll
  for (int i = 0; i < 8; ++i) z[i] = 0.0f;
  return z;
}

__device__ __forceinline__ float8 tile_mma(const half16* a, const half16* b) {
  float8 acc = zero8();
#pragma unroll
  for (int k = 0; k < KT; ++k)
    acc = __builtin_amdgcn_wmma_f32_16x16x32_f16(false, a[k], false, b[k],
                                                 (short)0, acc, false, false);
  return acc;
}

// ---- global -> LDS staging (16 bytes per call) -----------------------------
#if defined(HAVE_ASYNC_LDS)
// Builtin signature (from clang diagnostic): param0 is
// "__vector_size__(4*sizeof(int)) int __device__*"  (AS1 = __device__),
// param1 the AS3 (__shared__) v4i*.  Build exactly-typed pointers via integer
// casts: generic-shared low 32 bits are the LDS byte offset (aperture in the
// high word), so truncation yields the AS(3) representation.
typedef int v4i_ __attribute__((vector_size(16)));
typedef __attribute__((address_space(1))) v4i_* as1_v4i;
typedef __attribute__((address_space(3))) v4i_* as3_v4i;
#endif

__device__ __forceinline__ void copy_b128(const _Float16* __restrict__ g,
                                          _Float16* __restrict__ l) {
#if defined(HAVE_ASYNC_LDS)
  __builtin_amdgcn_global_load_async_to_lds_b128(
      (as1_v4i)(unsigned long long)(uintptr_t)g,
      (as3_v4i)(unsigned int)(uintptr_t)l, 0, 0);
#else
  *(half8*)l = *(const half8*)g;
#endif
}

// 256 threads stage one 16x256-half (8KB) tile into a padded LDS buffer.
__device__ __forceinline__ void stage_tile(const _Float16* __restrict__ g,
                                           _Float16* __restrict__ l, int tid) {
  int r = tid >> 4, c = tid & 15;                 // row 0..15, chunk 0..15
  const _Float16* gs = g + r * 256 + c * 8;       // 16B chunks, coalesced
  _Float16*       ls = l + r * LDS_STRIDE + c * 8;
  copy_b128(gs, ls);
  copy_b128(gs + 128, ls + 128);                  // second 256B half of row
}

__device__ __forceinline__ void stage_wait() {
#if defined(HAVE_ASYNC_LDS)
  asm volatile("s_wait_asynccnt 0x0" ::: "memory");
#endif
  __syncthreads();
}

// float atomic max via sign-aware integer atomics (native global atomics).
__device__ __forceinline__ void atomicMaxF(float* addr, float v) {
  if (v >= 0.0f) atomicMax((int*)addr, __float_as_int(v));
  else           atomicMin((unsigned int*)addr, __float_as_uint(v));
}

// ---- per-channel mean of gt ------------------------------------------------
__global__ void k_mean(const float* __restrict__ gt, float* __restrict__ mu) {
  int c = blockIdx.x;
  const float* p = gt + (size_t)c * S_DIM;
  float s = 0.0f;
  for (int i = threadIdx.x; i < S_DIM; i += blockDim.x) s += p[i];
  __shared__ float red[256];
  red[threadIdx.x] = s;
  __syncthreads();
  for (int off = 128; off > 0; off >>= 1) {
    if (threadIdx.x < off) red[threadIdx.x] += red[threadIdx.x + off];
    __syncthreads();
  }
  if (threadIdx.x == 0) mu[c] = red[0] / (float)S_DIM;
}

// ---- center, L2-normalize along C, emit row-major f16 features -------------
__global__ void k_norm(const float* __restrict__ I, const float* __restrict__ T,
                       const float* __restrict__ mu,
                       _Float16* __restrict__ Fi, _Float16* __restrict__ Ft) {
  int s = blockIdx.x, c = threadIdx.x;
  float m  = mu[c];
  float iv = I[(size_t)c * S_DIM + s] - m;
  float tv = T[(size_t)c * S_DIM + s] - m;
  __shared__ float r1[256], r2[256];
  r1[c] = iv * iv;
  r2[c] = tv * tv;
  __syncthreads();
  for (int off = 128; off > 0; off >>= 1) {
    if (c < off) { r1[c] += r1[c + off]; r2[c] += r2[c + off]; }
    __syncthreads();
  }
  float ni = 1.0f / fmaxf(sqrtf(r1[0]), 1e-12f);
  float nt = 1.0f / fmaxf(sqrtf(r2[0]), 1e-12f);
  Fi[(size_t)s * C_DIM + c] = (_Float16)(iv * ni);
  Ft[(size_t)s * C_DIM + c] = (_Float16)(tv * nt);
}

__global__ void k_init(float* rowm, float* rowz, float* colmax) {
  int i = blockIdx.x * blockDim.x + threadIdx.x;
  if (i < S_DIM) { rowm[i] = -3.0e38f; rowz[i] = 0.0f; colmax[i] = 0.0f; }
}

// ---- pass 1: row-max of cos.  Block = 128 i-rows (wave w owns rows
// i0..i0+15 in registers); j tiles streamed through double-buffered LDS. -----
__global__ void __launch_bounds__(256)
k_rowmax(const _Float16* __restrict__ Fi, const _Float16* __restrict__ Ft,
         float* __restrict__ rowm) {
  __shared__ _Float16 sb[2][LDS_TILE_H];
  int tid = threadIdx.x, lane = tid & 31, wave = tid >> 5;
  int i0 = blockIdx.x * 128 + wave * 16;
  half16 a[KT];
#pragma unroll
  for (int k = 0; k < KT; ++k) a[k] = load_a(Fi, i0, k * 32, lane);
  float mx[8];
#pragma unroll
  for (int r = 0; r < 8; ++r) mx[r] = -3.0e38f;

  int step = gridDim.y;
  int jt = blockIdx.y;
  stage_tile(Ft + (size_t)jt * TILE_H, sb[0], tid);
  int cur = 0;
  for (; jt < NT; jt += step) {
    stage_wait();                                  // sb[cur] ready, prev reads done
    int jn = jt + step;                            // block-uniform prefetch
    if (jn < NT) stage_tile(Ft + (size_t)jn * TILE_H, sb[cur ^ 1], tid);
    half16 b[KT];
#pragma unroll
    for (int k = 0; k < KT; ++k) b[k] = lds_b(sb[cur], k * 32, lane);
    float8 acc = tile_mma(a, b);
#pragma unroll
    for (int r = 0; r < 8; ++r) mx[r] = fmaxf(mx[r], acc[r]);
    cur ^= 1;
  }
#pragma unroll
  for (int r = 0; r < 8; ++r) {                    // reduce 16 columns per half
    float v = mx[r];
    for (int msk = 1; msk < 16; msk <<= 1) v = fmaxf(v, __shfl_xor(v, msk, 32));
    mx[r] = v;
  }
  if ((lane & 15) == 0) {
    int rbase = i0 + ((lane >> 4) << 3);
#pragma unroll
    for (int r = 0; r < 8; ++r) atomicMaxF(&rowm[rbase + r], mx[r]);
  }
}

// rowm -> exp scale  (dist_min = clip((1-maxcos)/2); scale = -2/(min+eps))
__global__ void k_scale(const float* __restrict__ rowm, float* __restrict__ scale) {
  int i = blockIdx.x * blockDim.x + threadIdx.x;
  if (i < S_DIM) {
    float m = fmaxf(0.0f, (1.0f - rowm[i]) * 0.5f);
    scale[i] = -2.0f / (m + 1e-5f);
  }
}

// ---- pass 2: Z_i = sum_j exp(scale_i * dist_ij) ----------------------------
__global__ void __launch_bounds__(256)
k_rowsum(const _Float16* __restrict__ Fi, const _Float16* __restrict__ Ft,
         const float* __restrict__ scale, float* __restrict__ rowz) {
  __shared__ _Float16 sb[2][LDS_TILE_H];
  int tid = threadIdx.x, lane = tid & 31, wave = tid >> 5;
  int i0 = blockIdx.x * 128 + wave * 16;
  int kh = lane >> 4;
  half16 a[KT];
#pragma unroll
  for (int k = 0; k < KT; ++k) a[k] = load_a(Fi, i0, k * 32, lane);
  float sr[8], sm[8];
#pragma unroll
  for (int r = 0; r < 8; ++r) { sr[r] = scale[i0 + (kh << 3) + r]; sm[r] = 0.0f; }

  int step = gridDim.y;
  int jt = blockIdx.y;
  stage_tile(Ft + (size_t)jt * TILE_H, sb[0], tid);
  int cur = 0;
  for (; jt < NT; jt += step) {
    stage_wait();
    int jn = jt + step;
    if (jn < NT) stage_tile(Ft + (size_t)jn * TILE_H, sb[cur ^ 1], tid);
    half16 b[KT];
#pragma unroll
    for (int k = 0; k < KT; ++k) b[k] = lds_b(sb[cur], k * 32, lane);
    float8 acc = tile_mma(a, b);
#pragma unroll
    for (int r = 0; r < 8; ++r) {
      float d = fmaxf(0.0f, (1.0f - acc[r]) * 0.5f);
      sm[r] += __expf(sr[r] * d);
    }
    cur ^= 1;
  }
#pragma unroll
  for (int r = 0; r < 8; ++r) {
    float v = sm[r];
    for (int msk = 1; msk < 16; msk <<= 1) v += __shfl_xor(v, msk, 32);
    sm[r] = v;
  }
  if ((lane & 15) == 0) {
    int rbase = i0 + (kh << 3);
#pragma unroll
    for (int r = 0; r < 8; ++r) atomicAdd(&rowz[rbase + r], sm[r]);
  }
}

// ---- pass 3: colmax_j = max_i exp(scale_i*dist)/Z_i.  Block = 128 j-cols
// (wave w owns cols in registers); i tiles streamed through LDS. -------------
__global__ void __launch_bounds__(256)
k_colmax(const _Float16* __restrict__ Fi, const _Float16* __restrict__ Ft,
         const float* __restrict__ scale, const float* __restrict__ rowz,
         float* __restrict__ colmax) {
  __shared__ _Float16 sa[2][LDS_TILE_H];
  int tid = threadIdx.x, lane = tid & 31, wave = tid >> 5;
  int j0 = blockIdx.x * 128 + wave * 16;
  int kh = lane >> 4;
  half16 b[KT];
#pragma unroll
  for (int k = 0; k < KT; ++k) b[k] = load_b(Ft, j0, k * 32, lane);
  float cm = 0.0f;

  int step = gridDim.y;
  int it = blockIdx.y;
  stage_tile(Fi + (size_t)it * TILE_H, sa[0], tid);
  int cur = 0;
  for (; it < NT; it += step) {
    stage_wait();
    int in2 = it + step;
    if (in2 < NT) stage_tile(Fi + (size_t)in2 * TILE_H, sa[cur ^ 1], tid);
    half16 a[KT];
#pragma unroll
    for (int k = 0; k < KT; ++k) a[k] = lds_a(sa[cur], k * 32, lane);
    float8 acc = tile_mma(a, b);
    int i0 = it * 16;
    const float* sp = scale + i0 + (kh << 3);
    const float* zp = rowz  + i0 + (kh << 3);
#pragma unroll
    for (int r = 0; r < 8; ++r) {
      float d = fmaxf(0.0f, (1.0f - acc[r]) * 0.5f);
      float u = __expf(sp[r] * d) / zp[r];
      cm = fmaxf(cm, u);
    }
    cur ^= 1;
  }
  cm = fmaxf(cm, __shfl_xor(cm, 16, 32));          // lane L vs L+16 same column
  if (lane < 16) atomicMaxF(&colmax[j0 + lane], cm);
}

// ---- final: loss = -log(mean_j colmax_j) -----------------------------------
__global__ void k_final(const float* __restrict__ colmax, float* __restrict__ out) {
  __shared__ float red[256];
  float s = 0.0f;
  for (int j = threadIdx.x; j < S_DIM; j += 256) s += colmax[j];
  red[threadIdx.x] = s;
  __syncthreads();
  for (int off = 128; off > 0; off >>= 1) {
    if (threadIdx.x < off) red[threadIdx.x] += red[threadIdx.x + off];
    __syncthreads();
  }
  if (threadIdx.x == 0) out[0] = -logf(red[0] / (float)S_DIM);
}

extern "C" void kernel_launch(void* const* d_in, const int* in_sizes, int n_in,
                              void* d_out, int out_size, void* d_ws, size_t ws_size,
                              hipStream_t stream) {
  (void)in_sizes; (void)n_in; (void)out_size; (void)ws_size;
  const float* images = (const float*)d_in[0];
  const float* gt     = (const float*)d_in[1];
  float* out          = (float*)d_out;

  // workspace carve-out (~9.6 MB)
  char* w = (char*)d_ws;
  float*    mu     = (float*)w;      w += 256 * sizeof(float);
  _Float16* Fi     = (_Float16*)w;   w += (size_t)S_DIM * C_DIM * sizeof(_Float16);
  _Float16* Ft     = (_Float16*)w;   w += (size_t)S_DIM * C_DIM * sizeof(_Float16);
  float*    rowm   = (float*)w;      w += S_DIM * sizeof(float);
  float*    scale  = (float*)w;      w += S_DIM * sizeof(float);
  float*    rowz   = (float*)w;      w += S_DIM * sizeof(float);
  float*    colmax = (float*)w;      w += S_DIM * sizeof(float);

  k_mean<<<C_DIM, 256, 0, stream>>>(gt, mu);
  k_norm<<<S_DIM, 256, 0, stream>>>(images, gt, mu, Fi, Ft);
  k_init<<<(S_DIM + 255) / 256, 256, 0, stream>>>(rowm, rowz, colmax);
  k_rowmax<<<dim3(NB, JCH), 256, 0, stream>>>(Fi, Ft, rowm);
  k_scale<<<(S_DIM + 255) / 256, 256, 0, stream>>>(rowm, scale);
  k_rowsum<<<dim3(NB, JCH), 256, 0, stream>>>(Fi, Ft, scale, rowz);
  k_colmax<<<dim3(NB, JCH), 256, 0, stream>>>(Fi, Ft, scale, rowz, colmax);
  k_final<<<1, 256, 0, stream>>>(colmax, out);
}
